// Encoder_1262720385203
// MI455X (gfx1250) — compile-verified, matrix operand order checked by reading, not run on hardware
//
#include <hip/hip_runtime.h>
#include <hip/hip_bf16.h>
#include <math.h>

// ---------------------------------------------------------------------------
// Bidirectional LSTM encoder for MI455X (gfx1250, wave32, WMMA bf16)
//   B=64, T=512, D=256, H=512, 4H=2048
// ---------------------------------------------------------------------------

typedef __attribute__((ext_vector_type(16))) __bf16 v16bf;
typedef __attribute__((ext_vector_type(8)))  float  v8f;
typedef __attribute__((ext_vector_type(4)))  float  f4;
typedef __attribute__((ext_vector_type(4)))  int    i4;

static constexpr int BB  = 64;    // batch
static constexpr int TT  = 512;   // time
static constexpr int DD  = 256;   // input dim
static constexpr int HH  = 512;   // hidden dim
static constexpr int N4H = 2048;  // 4*H

__device__ __forceinline__ float sigm(float x) {
    return 1.0f / (1.0f + __expf(-x));
}

// ---- optional CDNA5 async global->LDS path (ASYNCcnt) --------------------
#if __has_builtin(__builtin_amdgcn_global_load_async_to_lds_b128) && \
    __has_builtin(__builtin_amdgcn_s_wait_asynccnt)
#define ASYNC_LDS 1
__device__ __forceinline__ void async_cp16(void* lds, const void* g) {
    __builtin_amdgcn_global_load_async_to_lds_b128(
        (__attribute__((address_space(1))) i4*)(i4*)(void*)g,
        (__attribute__((address_space(3))) i4*)(i4*)lds, 0, 0);
}
__device__ __forceinline__ void async_wait0() {
    __builtin_amdgcn_s_wait_asynccnt(0);
}
#endif

// ---------------------------------------------------------------------------
// init: reset grid-barrier counters (must be deterministic across replays)
// ---------------------------------------------------------------------------
__global__ void k_init(unsigned* cnt) {
    if (threadIdx.x < 2) cnt[threadIdx.x] = 0u;
}

// ---------------------------------------------------------------------------
// convert x [B,T,D] f32  ->  xb [T*B, D] bf16   (row = t*B + b)
// ---------------------------------------------------------------------------
__global__ void k_cvt_x(const float* __restrict__ x, __bf16* __restrict__ xb) {
    size_t i = (size_t)blockIdx.x * blockDim.x + threadIdx.x;
    int d   = (int)(i % DD);
    int row = (int)(i / DD);
    int t = row / BB, b = row % BB;
    xb[i] = (__bf16)x[((size_t)b * TT + t) * DD + d];
}

// ---------------------------------------------------------------------------
// transpose+convert: W [K,N] f32 -> Wt [N,K] bf16
// ---------------------------------------------------------------------------
__global__ void k_tcvt(const float* __restrict__ w, __bf16* __restrict__ wt,
                       int K, int N) {
    size_t i = (size_t)blockIdx.x * blockDim.x + threadIdx.x;
    int k = (int)(i % K);
    int n = (int)(i / K);
    wt[i] = (__bf16)w[(size_t)k * N + n];
}

// ---------------------------------------------------------------------------
// GEMM: C[M,2048] = A[M,256](bf16) @ Wt[2048,256](bf16,row=n) + bias
// block = 256 thr (8 waves), tile 128x128, bf16 WMMA 16x16x32, fp32 acc
// ---------------------------------------------------------------------------
__global__ __launch_bounds__(256)
void k_gemm_xw(const __bf16* __restrict__ A,
               const __bf16* __restrict__ Bt,
               const float*  __restrict__ bias,
               float* __restrict__ C) {
    constexpr int K   = DD;   // 256
    constexpr int LDA = 40;   // padded bf16 stride for 32-wide k-chunk
    __shared__ __bf16 sA[128 * LDA];
    __shared__ __bf16 sB[128 * LDA];

    const int tid  = threadIdx.x;
    const int wave = tid >> 5;
    const int lane = tid & 31;
    const int half = lane >> 4;
    const int l16  = lane & 15;
    const int m_base = blockIdx.y * 128;
    const int n_base = blockIdx.x * 128;
    const int wm = (wave & 1) * 64;   // wave row offset
    const int wn = (wave >> 1) * 32;  // wave col offset

    v8f acc[4][2];
    for (int mt = 0; mt < 4; ++mt)
        for (int nt = 0; nt < 2; ++nt)
            acc[mt][nt] = (v8f)(0.0f);

    const int crow  = tid >> 1;        // 0..127
    const int chalf = (tid & 1) * 16;  // element offset 0 / 16
    const int k0    = half * 8;

    for (int kk = 0; kk < K; kk += 32) {
        // stage A/B 128x32 bf16 tiles
        const __bf16* gA = &A [(size_t)(m_base + crow) * K + kk + chalf];
        const __bf16* gB = &Bt[(size_t)(n_base + crow) * K + kk + chalf];
        __bf16* lA = &sA[crow * LDA + chalf];
        __bf16* lB = &sB[crow * LDA + chalf];
#ifdef ASYNC_LDS
        async_cp16(lA,     gA);
        async_cp16(lA + 8, gA + 8);
        async_cp16(lB,     gB);
        async_cp16(lB + 8, gB + 8);
        async_wait0();
#else
        ((f4*)lA)[0] = ((const f4*)gA)[0];
        ((f4*)lA)[1] = ((const f4*)gA)[1];
        ((f4*)lB)[0] = ((const f4*)gB)[0];
        ((f4*)lB)[1] = ((const f4*)gB)[1];
#endif
        __syncthreads();

        union Frag { v16bf v; f4 q[2]; };
        Frag af[4], bfr[2];
        for (int mt = 0; mt < 4; ++mt) {
            const __bf16* p = &sA[(wm + mt * 16 + l16) * LDA + k0];
            af[mt].q[0] = *(const f4*)p;
            af[mt].q[1] = *(const f4*)(p + 16);
        }
        for (int nt = 0; nt < 2; ++nt) {
            const __bf16* p = &sB[(wn + nt * 16 + l16) * LDA + k0];
            bfr[nt].q[0] = *(const f4*)p;
            bfr[nt].q[1] = *(const f4*)(p + 16);
        }
        for (int mt = 0; mt < 4; ++mt)
            for (int nt = 0; nt < 2; ++nt)
                acc[mt][nt] = __builtin_amdgcn_wmma_f32_16x16x32_bf16(
                    false, af[mt].v, false, bfr[nt].v,
                    (short)0, acc[mt][nt], false, false);
        __syncthreads();
    }

    // epilogue: add bias, store fp32
    for (int nt = 0; nt < 2; ++nt) {
        const int col = n_base + wn + nt * 16 + l16;
        const float bv = bias[col];
        for (int mt = 0; mt < 4; ++mt) {
            const int row0 = m_base + wm + mt * 16 + half * 8;
            union { v8f v; float f[8]; } u; u.v = acc[mt][nt];
            for (int r = 0; r < 8; ++r)
                C[(size_t)(row0 + r) * N4H + col] = u.f[r] + bv;
        }
    }
}

// ---------------------------------------------------------------------------
// Persistent recurrent kernel: 16 blocks x 256 thr, grid barrier per step.
// Block bk owns hidden columns [bk*32, bk*32+32) -> 128 gate cols of z.
// z = xW[t] + h @ Ut ; gates -> c,h update ; h broadcast via global bf16 buf.
// ---------------------------------------------------------------------------
template <int REV, int ACCUM>
__global__ __launch_bounds__(256)
void k_lstm(const float*  __restrict__ xW,    // [T*B, 2048] fp32
            const __bf16* __restrict__ Ut,    // [2048, 512] bf16 (row = z-col)
            __bf16*       __restrict__ hg,    // [64, 512] bf16 h broadcast buf
            float*        __restrict__ out,   // [B, T, H] fp32
            float*        __restrict__ h_last,// [64, 512]
            float*        __restrict__ c_last,// [64, 512]
            unsigned*     __restrict__ bar) {
    constexpr int LDH = 520;  // padded bf16 stride (banks)
    constexpr int LDZ = 132;  // padded f32 stride
    __shared__ __bf16 sH[64 * LDH];  // 66,560 B : h as WMMA A-matrix
    __shared__ float  sZ[64 * LDZ];  // 33,792 B : z exchange between waves

    const int tid  = threadIdx.x;
    const int wave = tid >> 5;
    const int lane = tid & 31;
    const int half = lane >> 4;
    const int l16  = lane & 15;
    const int jb   = blockIdx.x * 32;   // hidden-col base of this block
    const int wm   = (wave >> 2) * 32;  // wave row offset (0 / 32)
    const int wnp  = wave & 3;          // gate index owned by this wave
    const int k0   = half * 8;

    // elementwise ownership: thread -> (batch row eb, hidden cols eu..eu+7)
    const int eb = tid >> 2;
    const int eu = (tid & 3) * 8;
    float c[8];
    for (int e = 0; e < 8; ++e) c[e] = 0.0f;

    for (int step = 0; step < TT; ++step) {
        const int t = REV ? (TT - 1 - step) : step;

        if (step > 0) {
            // stage h (written by all blocks last step) into LDS
#ifdef ASYNC_LDS
            for (int i = 0; i < 16; ++i) {
                const int cid = tid + 256 * i;      // 0..4095 chunks of 8 bf16
                const int row = cid >> 6;
                const int col = (cid & 63) * 8;
                async_cp16(&sH[row * LDH + col], &hg[row * HH + col]);
            }
            async_wait0();
#else
            for (int i = 0; i < 16; ++i) {
                const int cid = tid + 256 * i;
                const int row = cid >> 6;
                const int col = (cid & 63) * 8;
                *(f4*)&sH[row * LDH + col] = *(const f4*)&hg[row * HH + col];
            }
#endif
        }

        // init accumulators with xW[t] slice (gate pre-activations)
        v8f acc[2][2];
        const float* xwt = xW + (size_t)t * BB * N4H;
        for (int ni = 0; ni < 2; ++ni) {
            const int zcol = wnp * HH + jb + ni * 16 + l16;
            for (int mi = 0; mi < 2; ++mi) {
                const int row0 = wm + mi * 16 + half * 8;
                union { v8f v; float f[8]; } u;
                for (int r = 0; r < 8; ++r)
                    u.f[r] = xwt[(size_t)(row0 + r) * N4H + zcol];
                acc[mi][ni] = u.v;
            }
        }

        // speculative prefetch of next step's xW slice (one 128B line/thread)
        if (step < TT - 1) {
            const int tn = REV ? (t - 1) : (t + 1);
            const float* pf =
                &xW[(size_t)tn * BB * N4H + (size_t)eb * N4H + (tid & 3) * HH + jb];
            __builtin_prefetch(pf, 0, 1);
        }

        if (step > 0) {
            __syncthreads();
            for (int kk = 0; kk < HH; kk += 32) {
                union Frag { v16bf v; f4 q[2]; };
                Frag af[2], bfr[2];
                for (int mi = 0; mi < 2; ++mi) {
                    const __bf16* p = &sH[(wm + mi * 16 + l16) * LDH + kk + k0];
                    af[mi].q[0] = *(const f4*)p;
                    af[mi].q[1] = *(const f4*)(p + 16);
                }
                for (int ni = 0; ni < 2; ++ni) {
                    const int zn = wnp * HH + jb + ni * 16 + l16;
                    const __bf16* p = &Ut[(size_t)zn * HH + kk + k0];
                    bfr[ni].q[0] = *(const f4*)p;
                    bfr[ni].q[1] = *(const f4*)(p + 16);
                }
                for (int mi = 0; mi < 2; ++mi)
                    for (int ni = 0; ni < 2; ++ni)
                        acc[mi][ni] = __builtin_amdgcn_wmma_f32_16x16x32_bf16(
                            false, af[mi].v, false, bfr[ni].v,
                            (short)0, acc[mi][ni], false, false);
            }
        }

        // publish z tiles to LDS for cross-wave gate combine
        for (int ni = 0; ni < 2; ++ni) {
            const int cl = wnp * 32 + ni * 16 + l16;  // local z col 0..127
            for (int mi = 0; mi < 2; ++mi) {
                const int row0 = wm + mi * 16 + half * 8;
                union { v8f v; float f[8]; } u; u.v = acc[mi][ni];
                for (int r = 0; r < 8; ++r)
                    sZ[(row0 + r) * LDZ + cl] = u.f[r];
            }
        }
        __syncthreads();

        // gates (all sigmoid per reference), c/h update
        union { __bf16 hb[8]; f4 q; } hpack;
        float hv[8];
        const float* zr = &sZ[eb * LDZ];
        for (int e = 0; e < 8; ++e) {
            const int u = eu + e;
            const float ig = sigm(zr[u]);
            const float fg = sigm(zr[32 + u]);
            const float gg = sigm(zr[64 + u]);
            const float og = sigm(zr[96 + u]);
            const float cn = fg * c[e] + ig * gg;
            c[e] = cn;
            const float h = og * sigm(cn);
            hv[e] = h;
            hpack.hb[e] = (__bf16)h;
        }

        // h for next step (bf16 broadcast buffer)
        *(f4*)&hg[eb * HH + jb + eu] = hpack.q;

        // output: fwd writes 0.5*h, bwd accumulates
        float* op = &out[((size_t)eb * TT + t) * HH + jb + eu];
        for (int e = 0; e < 8; ++e) {
            if (ACCUM) op[e] += 0.5f * hv[e];
            else       op[e]  = 0.5f * hv[e];
        }

        if (step == TT - 1) {
            for (int e = 0; e < 8; ++e) {
                h_last[eb * HH + jb + eu + e] = hv[e];
                c_last[eb * HH + jb + eu + e] = c[e];
            }
        }

        // device-scope grid barrier (monotonic counter, reset each launch)
        if (step < TT - 1) {
            __threadfence();
            __syncthreads();
            if (tid == 0) {
                __hip_atomic_fetch_add(bar, 1u, __ATOMIC_RELEASE,
                                       __HIP_MEMORY_SCOPE_AGENT);
                const unsigned target = (unsigned)(step + 1) * gridDim.x;
                while (__hip_atomic_load(bar, __ATOMIC_ACQUIRE,
                                         __HIP_MEMORY_SCOPE_AGENT) < target) {
                    __builtin_amdgcn_s_sleep(2);
                }
            }
            __syncthreads();
        }
    }
}

// ---------------------------------------------------------------------------
// launch
// ---------------------------------------------------------------------------
extern "C" void kernel_launch(void* const* d_in, const int* in_sizes, int n_in,
                              void* d_out, int out_size, void* d_ws,
                              size_t ws_size, hipStream_t stream) {
    const float* x   = (const float*)d_in[0];
    // d_in[1] = 'hidden' : unused by the reference
    const float* Wf  = (const float*)d_in[2];
    const float* Uf  = (const float*)d_in[3];
    const float* bf_ = (const float*)d_in[4];
    const float* Wb  = (const float*)d_in[5];
    const float* Ub  = (const float*)d_in[6];
    const float* bb_ = (const float*)d_in[7];

    float* out = (float*)d_out;
    float* hF  = out + (size_t)BB * TT * HH;
    float* cF  = hF + (size_t)BB * HH;
    float* hB  = cF + (size_t)BB * HH;
    float* cB  = hB + (size_t)BB * HH;

    // workspace carve-up (256B aligned)
    char* ws = (char*)d_ws;
    size_t off = 0;
    auto carve = [&](size_t bytes) {
        void* p = ws + off;
        off = (off + bytes + 255) & ~(size_t)255;
        return p;
    };
    unsigned* cnt   = (unsigned*)carve(256);
    __bf16*   xb    = (__bf16*)  carve((size_t)TT * BB * DD * 2);       // 16 MB
    __bf16*   Wtf   = (__bf16*)  carve((size_t)N4H * DD * 2);           //  1 MB
    __bf16*   Wtb   = (__bf16*)  carve((size_t)N4H * DD * 2);           //  1 MB
    __bf16*   Utf   = (__bf16*)  carve((size_t)N4H * HH * 2);           //  2 MB
    __bf16*   Utb   = (__bf16*)  carve((size_t)N4H * HH * 2);           //  2 MB
    __bf16*   hFbuf = (__bf16*)  carve((size_t)BB * HH * 2);            // 64 KB
    __bf16*   hBbuf = (__bf16*)  carve((size_t)BB * HH * 2);            // 64 KB
    float*    xWf   = (float*)   carve((size_t)TT * BB * N4H * 4);      // 256 MB
    float*    xWb   = (float*)   carve((size_t)TT * BB * N4H * 4);      // 256 MB

    k_init<<<1, 64, 0, stream>>>(cnt);

    // precision staging
    k_cvt_x<<<(TT * BB * DD) / 256, 256, 0, stream>>>(x, xb);
    k_tcvt<<<(DD * N4H) / 256, 256, 0, stream>>>(Wf, Wtf, DD, N4H);
    k_tcvt<<<(DD * N4H) / 256, 256, 0, stream>>>(Wb, Wtb, DD, N4H);
    k_tcvt<<<(HH * N4H) / 256, 256, 0, stream>>>(Uf, Utf, HH, N4H);
    k_tcvt<<<(HH * N4H) / 256, 256, 0, stream>>>(Ub, Utb, HH, N4H);

    // input projections: xW = x @ W + b (bf16 WMMA, fp32 out)
    dim3 ggrid(N4H / 128, (TT * BB) / 128);
    k_gemm_xw<<<ggrid, 256, 0, stream>>>(xb, Wtf, bf_, xWf);
    k_gemm_xw<<<ggrid, 256, 0, stream>>>(xb, Wtb, bb_, xWb);

    // recurrence: forward then backward (backward accumulates into out)
    k_lstm<0, 0><<<16, 256, 0, stream>>>(xWf, Utf, hFbuf, out, hF, cF, cnt + 0);
    k_lstm<1, 1><<<16, 256, 0, stream>>>(xWb, Utb, hBbuf, out, hB, cB, cnt + 1);
}
